// MHAttention_18004502905182
// MI455X (gfx1250) — matrix operand order repaired and checked by reading, not compiled, not run on hardware
//
#include <hip/hip_runtime.h>

// ---------------------------------------------------------------------------
// MI455X (gfx1250) fused multi-head attention + residual + LayerNorm.
// bf16 WMMA (v_wmma_f32_16x16x32_bf16) with f32 accumulation everywhere.
// Flash-attention kernel stages K/V tiles into LDS with the Tensor Data
// Mover (tensor_load_to_lds + s_wait_tensorcnt), double buffered.
// ---------------------------------------------------------------------------

typedef __attribute__((ext_vector_type(16))) __bf16 v16bf;
typedef __attribute__((ext_vector_type(8)))  __bf16 v8bf;
typedef __attribute__((ext_vector_type(8)))  float  v8f;
typedef __attribute__((ext_vector_type(4)))  unsigned int u32x4;
typedef __attribute__((ext_vector_type(8)))  int   i32x8;
typedef __attribute__((ext_vector_type(4)))  int   i32x4;

#define BATCH   4
#define SEQ     2048
#define DMODEL  1024
#define HEADS   16
#define DH      64
#define ROWS    (BATCH * SEQ)          // 8192 token rows

#if defined(__has_builtin)
#if __has_builtin(__builtin_amdgcn_tensor_load_to_lds)
#define USE_TDM 1
#endif
#endif

__device__ __forceinline__ int lane_id() { return (int)(threadIdx.x & 31); }

__device__ __forceinline__ v8f wmma_bf16(v16bf a, v16bf b, v8f c) {
  // (neg_a, A, neg_b, B, c_mod, C, reuse_a, reuse_b)
  return __builtin_amdgcn_wmma_f32_16x16x32_bf16(false, a, false, b, (short)0, c,
                                                 false, false);
}

// A-matrix 16x32 (16-bit) tile load, ISA §7.12.2 layout:
//   lanes 0-15 : M=lane,    VGPR0-3 K=0..7,  VGPR4-7 K=16..23
//   lanes 16-31: M=lane-16, VGPR0-3 K=8..15, VGPR4-7 K=24..31
__device__ __forceinline__ v16bf load_tile_a(const __bf16* p, int ld) {
  const int lane = lane_id();
  const int r    = lane & 15;
  const int khi  = (lane >> 4) << 3;           // 0 or 8
  v8bf lo = *reinterpret_cast<const v8bf*>(p + (size_t)r * ld + khi);
  v8bf hi = *reinterpret_cast<const v8bf*>(p + (size_t)r * ld + 16 + khi);
  return __builtin_shufflevector(lo, hi, 0,1,2,3,4,5,6,7,8,9,10,11,12,13,14,15);
}

// B-matrix 32x16 (16-bit) tile load (same K order as A):
//   lanes 0-15: N=lane, K=0..7,16..23; lanes 16-31: N=lane-16, K=8..15,24..31.
// p points at a K-contiguous layout: row n holds the K values, ld elems.
__device__ __forceinline__ v16bf load_tile_b(const __bf16* p, int ld) {
  const int lane = lane_id();
  const __bf16* q = p + (size_t)(lane & 15) * ld + ((lane >> 4) << 3);
  v8bf lo = *reinterpret_cast<const v8bf*>(q);
  v8bf hi = *reinterpret_cast<const v8bf*>(q + 16);
  return __builtin_shufflevector(lo, hi, 0,1,2,3,4,5,6,7,8,9,10,11,12,13,14,15);
}

// ---------------------------------------------------------------------------
// Tensor Data Mover: 2-D tile load global -> LDS (D# per CDNA5 ISA §8).
// All sizes/strides are in elements (data_size = 2 bytes).
// ---------------------------------------------------------------------------
__device__ __forceinline__ unsigned lds_off(const void* p) {
  return (unsigned)(size_t)p;                  // low 32 bits = LDS byte offset
}

#ifdef USE_TDM
__device__ __forceinline__ void tdm_load_2d(unsigned lds_addr, const void* gp,
                                            unsigned tile_d0, unsigned tile_d1,
                                            unsigned long long tensor_d0,
                                            unsigned long long tensor_d1,
                                            unsigned long long stride0) {
  const unsigned long long ga = (unsigned long long)(size_t)gp;
  u32x4 g0;
  g0[0] = 1u;                                          // count=1, user mode
  g0[1] = lds_addr;                                    // lds_addr [63:32]
  g0[2] = (unsigned)(ga & 0xffffffffull);              // global_addr [95:64]
  g0[3] = (unsigned)((ga >> 32) & 0x01ffffffull) | (2u << 30);  // type=2
  i32x8 g1;
  g1[0] = (int)(1u << 16);                             // data_size=1 (2B)
  g1[1] = (int)((unsigned)(tensor_d0 & 0xffffull) << 16);
  g1[2] = (int)(((unsigned)(tensor_d0 >> 16) & 0xffffu) |
                ((unsigned)(tensor_d1 & 0xffffull) << 16));
  g1[3] = (int)(((unsigned)(tensor_d1 >> 16) & 0xffffu) | (tile_d0 << 16));
  g1[4] = (int)(tile_d1 & 0xffffu);                    // tile_dim1, tile_dim2=0
  g1[5] = (int)(unsigned)(stride0 & 0xffffffffull);    // dim0 stride [31:0]
  g1[6] = (int)((unsigned)((stride0 >> 32) & 0xffffull));
  g1[7] = 0;
  const i32x4 z4 = {0, 0, 0, 0};
#if __clang_major__ >= 23
  const i32x8 z8 = {0, 0, 0, 0, 0, 0, 0, 0};
  __builtin_amdgcn_tensor_load_to_lds(g0, g1, z4, z4, z8, 0);
#else
  __builtin_amdgcn_tensor_load_to_lds(g0, g1, z4, z4, 0);
#endif
}
#endif

// Stage one Sk-chunk (64 keys) of K and V into LDS.
//   kdst: [64 s][64 dh] row-major;  vdst: [64 dh][64 s] row-major.
__device__ __forceinline__ void stage_chunk(__bf16* kdst, __bf16* vdst,
                                            const __bf16* kp, const __bf16* vp,
                                            int kb, int wave) {
#ifdef USE_TDM
  if (wave == 0) {
    tdm_load_2d(lds_off(kdst), kp + (size_t)kb * DH,
                /*tile_d0=*/DH, /*tile_d1=*/64,
                /*tensor_d0=*/DH, /*tensor_d1=*/SEQ, /*stride0=*/DH);
    tdm_load_2d(lds_off(vdst), vp + kb,
                /*tile_d0=*/64, /*tile_d1=*/DH,
                /*tensor_d0=*/SEQ, /*tensor_d1=*/DH, /*stride0=*/SEQ);
  }
#else
  // Cooperative fallback: K chunk is one contiguous 8KB span.
  for (int i = threadIdx.x; i < (64 * DH) / 8; i += 256)
    reinterpret_cast<v8bf*>(kdst)[i] =
        reinterpret_cast<const v8bf*>(kp + (size_t)kb * DH)[i];
  for (int i = threadIdx.x; i < (DH * 64) / 8; i += 256) {
    const int d = i >> 3, c8 = (i & 7) << 3;
    *reinterpret_cast<v8bf*>(vdst + d * 64 + c8) =
        *reinterpret_cast<const v8bf*>(vp + (size_t)d * SEQ + kb + c8);
  }
#endif
}

__device__ __forceinline__ void wait_tensor0() {
#ifdef USE_TDM
#if __has_builtin(__builtin_amdgcn_s_wait_tensorcnt)
  __builtin_amdgcn_s_wait_tensorcnt(0);
#else
  asm volatile("s_wait_tensorcnt 0" ::: "memory");
#endif
#endif
}

// ---------------------------------------------------------------------------
// Kernel 0: fp32 -> bf16 elementwise convert
// ---------------------------------------------------------------------------
__global__ void __launch_bounds__(256) cvt_kernel(const float* __restrict__ in,
                                                  __bf16* __restrict__ out, int n) {
  int i = blockIdx.x * 256 + threadIdx.x;
  if (i < n) out[i] = (__bf16)in[i];
}

// Kernel 0b: fp32 [1024,1024] row-major -> bf16 transposed [1024,1024]
__global__ void __launch_bounds__(256) tcvt_kernel(const float* __restrict__ in,
                                                   __bf16* __restrict__ out) {
  int i = blockIdx.x * 256 + threadIdx.x;          // over 1M elements
  int c = i >> 10;                                 // output row  (N)
  int r = i & 1023;                                // output col  (K)
  out[(size_t)c * 1024 + r] = (__bf16)in[(size_t)r * 1024 + c];
}

// ---------------------------------------------------------------------------
// Kernel 1: QKV projection.  grid = (ROWS/16, 24), block = 256 (8 waves)
// Q,K -> [B,H,S,dh] bf16.   V -> transposed [B,H,dh,S] bf16.
// ---------------------------------------------------------------------------
__global__ void __launch_bounds__(256) qkv_kernel(
    const __bf16* __restrict__ Xb,        // [8192,1024]
    const __bf16* __restrict__ WqT,       // [1024(N),1024(K)]
    const __bf16* __restrict__ WkT,
    const __bf16* __restrict__ WvT,
    const float* __restrict__ bq, const float* __restrict__ bk,
    const float* __restrict__ bv,
    __bf16* __restrict__ Qb, __bf16* __restrict__ Kb, __bf16* __restrict__ Vt) {
  const int lane    = lane_id();
  const int wave    = threadIdx.x >> 5;
  const int row0    = blockIdx.x * 16;
  const int mat     = blockIdx.y >> 3;                     // 0:Q 1:K 2:V
  const int colbase = (blockIdx.y & 7) * 128 + wave * 16;

  const __bf16* Wt   = (mat == 0) ? WqT : (mat == 1) ? WkT : WvT;
  const float*  bias = (mat == 0) ? bq  : (mat == 1) ? bk  : bv;

  v8f acc = {};
#pragma unroll 4
  for (int kc = 0; kc < DMODEL; kc += 32) {
    v16bf a = load_tile_a(Xb + (size_t)row0 * DMODEL + kc, DMODEL);
    v16bf b = load_tile_b(Wt + (size_t)colbase * DMODEL + kc, DMODEL);
    acc = wmma_bf16(a, b, acc);
  }

  const int nc    = lane & 15;
  const int rhalf = (lane >> 4) << 3;
  const int col   = colbase + nc;
  const float bb  = bias[col];
  const int b_    = row0 >> 11;
  const int h     = col >> 6;
  const int d     = col & 63;

  if (mat == 2) {
    const int s0 = (row0 & (SEQ - 1)) + rhalf;
    v8bf pack;
#pragma unroll
    for (int v = 0; v < 8; ++v) pack[v] = (__bf16)(acc[v] + bb);
    *reinterpret_cast<v8bf*>(Vt + ((((size_t)b_ * HEADS + h) * DH + d) * SEQ + s0)) = pack;
  } else {
    __bf16* dst = (mat == 0) ? Qb : Kb;
#pragma unroll
    for (int v = 0; v < 8; ++v) {
      const int s = (row0 & (SEQ - 1)) + rhalf + v;
      dst[(((size_t)b_ * HEADS + h) * SEQ + s) * DH + d] = (__bf16)(acc[v] + bb);
    }
  }
}

// ---------------------------------------------------------------------------
// Kernel 2: flash attention.  grid = 1024 blocks, 256 threads (8 waves).
// All 8 waves of a block share one (b,h); K/V Sk-chunks of 64 are staged into
// double-buffered LDS via TDM and consumed by WMMA from LDS.
// ---------------------------------------------------------------------------
__global__ void __launch_bounds__(256) attn_kernel(
    const __bf16* __restrict__ Qb,   // [B,H,S,dh]
    const __bf16* __restrict__ Kb,   // [B,H,S,dh]
    const __bf16* __restrict__ Vt,   // [B,H,dh,S]
    const float* __restrict__ mask,  // [B,1,S,S]
    __bf16* __restrict__ Ctx) {      // [B,S,H*dh]
  __shared__ alignas(32) __bf16 kbuf[2][64 * DH];   // [s_local][dh]   8KB each
  __shared__ alignas(32) __bf16 vbuf[2][DH * 64];   // [d][s_local]    8KB each
  __shared__ alignas(32) __bf16 pbuf[8][16 * 64];   // P tile per wave 16KB

  const int lane = lane_id();
  const int wave = threadIdx.x >> 5;
  const int wid  = blockIdx.x * 8 + wave;           // q-tile id
  const int qt   = wid & 127;
  const int bh   = wid >> 7;                        // uniform per block
  const int b_   = bh >> 4;
  const int h    = bh & 15;
  const int q0   = qt * 16;

  const __bf16* qp = Qb + ((size_t)bh * SEQ + q0) * DH;
  const __bf16* kp = Kb + (size_t)bh * SEQ * DH;
  const __bf16* vp = Vt + (size_t)bh * DH * SEQ;
  const float*  mp = mask + ((size_t)b_ * SEQ + q0) * SEQ;

  const v16bf aq0 = load_tile_a(qp, DH);
  const v16bf aq1 = load_tile_a(qp + 32, DH);

  v8f o[4];
#pragma unroll
  for (int t = 0; t < 4; ++t) o[t] = (v8f){};
  float m_[8], l_[8];
#pragma unroll
  for (int v = 0; v < 8; ++v) { m_[v] = -1e30f; l_[v] = 0.0f; }

  const int rhalf = (lane >> 4) << 3;
  const int nc    = lane & 15;

  stage_chunk(kbuf[0], vbuf[0], kp, vp, 0, wave);   // prologue: chunk 0

  for (int kb = 0; kb < SEQ; kb += 64) {
    const int buf = (kb >> 6) & 1;
    if (wave == 0) wait_tensor0();                  // chunk `buf` landed
    __syncthreads();                                // publish + all reads done
    if (kb + 64 < SEQ)
      stage_chunk(kbuf[buf ^ 1], vbuf[buf ^ 1], kp, vp, kb + 64, wave);

    // ---- scores: 4 x (16x16) C tiles over Sk=64, K-dim = dh = 64 ----
    v8f s[4];
#pragma unroll
    for (int t = 0; t < 4; ++t) {
      s[t] = (v8f){};
      v16bf b0 = load_tile_b(kbuf[buf] + (t * 16) * DH, DH);
      v16bf b1 = load_tile_b(kbuf[buf] + (t * 16) * DH + 32, DH);
      s[t] = wmma_bf16(aq0, b0, s[t]);
      s[t] = wmma_bf16(aq1, b1, s[t]);
    }

    // ---- scale + additive mask + online softmax (per row) ----
#pragma unroll
    for (int v = 0; v < 8; ++v) {
      const int qr = rhalf + v;
      float sv[4];
#pragma unroll
      for (int t = 0; t < 4; ++t)
        sv[t] = s[t][v] * 0.125f + mp[(size_t)qr * SEQ + kb + t * 16 + nc];

      float mx = fmaxf(fmaxf(sv[0], sv[1]), fmaxf(sv[2], sv[3]));
      mx = fmaxf(mx, __shfl_xor(mx, 1, 16));
      mx = fmaxf(mx, __shfl_xor(mx, 2, 16));
      mx = fmaxf(mx, __shfl_xor(mx, 4, 16));
      mx = fmaxf(mx, __shfl_xor(mx, 8, 16));

      const float mnew  = fmaxf(m_[v], mx);
      const float alpha = __expf(m_[v] - mnew);
      float p[4], rs = 0.0f;
#pragma unroll
      for (int t = 0; t < 4; ++t) { p[t] = __expf(sv[t] - mnew); rs += p[t]; }
      rs += __shfl_xor(rs, 1, 16);
      rs += __shfl_xor(rs, 2, 16);
      rs += __shfl_xor(rs, 4, 16);
      rs += __shfl_xor(rs, 8, 16);

      l_[v] = l_[v] * alpha + rs;
      m_[v] = mnew;
#pragma unroll
      for (int t = 0; t < 4; ++t) o[t][v] *= alpha;
#pragma unroll
      for (int t = 0; t < 4; ++t)
        pbuf[wave][qr * 64 + t * 16 + nc] = (__bf16)p[t];
    }
    // same-wave LDS RAW: P stores must land before the A-tile reload
    asm volatile("s_wait_dscnt 0" ::: "memory");

    // ---- PV: P(16x64) x V-chunk(64x64) -> 4 O tiles ----
    const v16bf ap0 = load_tile_a(&pbuf[wave][0], 64);
    const v16bf ap1 = load_tile_a(&pbuf[wave][0] + 32, 64);
#pragma unroll
    for (int t = 0; t < 4; ++t) {
      o[t] = wmma_bf16(ap0, load_tile_b(vbuf[buf] + (t * 16) * 64, 64), o[t]);
      o[t] = wmma_bf16(ap1, load_tile_b(vbuf[buf] + (t * 16) * 64 + 32, 64), o[t]);
    }
  }

  // ---- epilogue: O /= l, write ctx [B,S,H*dh] bf16 ----
#pragma unroll
  for (int v = 0; v < 8; ++v) {
    const float rcp = 1.0f / l_[v];
    const int   s   = q0 + rhalf + v;
    const size_t ro = ((size_t)b_ * SEQ + s) * (HEADS * DH) + h * DH;
#pragma unroll
    for (int t = 0; t < 4; ++t)
      Ctx[ro + t * 16 + nc] = (__bf16)(o[t][v] * rcp);
  }
}

// ---------------------------------------------------------------------------
// Kernel 3: output projection + bias + residual + LayerNorm (fused).
// grid = ROWS/16 = 512 blocks, 256 threads.
// ---------------------------------------------------------------------------
__global__ void __launch_bounds__(256) oproj_ln_kernel(
    const __bf16* __restrict__ Ctx,   // [8192,1024]
    const __bf16* __restrict__ WoT,   // [1024(N),1024(K)]
    const float* __restrict__ bo, const float* __restrict__ Xf,
    const float* __restrict__ gamma, const float* __restrict__ beta,
    float* __restrict__ Out) {
  __shared__ float tile[16][DMODEL];                // 64 KB staging

  const int lane = lane_id();
  const int wave = threadIdx.x >> 5;
  const int row0 = blockIdx.x * 16;
  const int nb   = wave * 128;

  v8f c[8];
#pragma unroll
  for (int t = 0; t < 8; ++t) c[t] = (v8f){};

  for (int kc = 0; kc < DMODEL; kc += 32) {
    v16bf a = load_tile_a(Ctx + (size_t)row0 * DMODEL + kc, DMODEL);
#pragma unroll
    for (int t = 0; t < 8; ++t) {
      v16bf b = load_tile_b(WoT + (size_t)(nb + t * 16) * DMODEL + kc, DMODEL);
      c[t] = wmma_bf16(a, b, c[t]);
    }
  }

  const int rhalf = (lane >> 4) << 3;
  const int nc    = lane & 15;
#pragma unroll
  for (int t = 0; t < 8; ++t) {
    const int col = nb + t * 16 + nc;
    const float bb = bo[col];
#pragma unroll
    for (int v = 0; v < 8; ++v) {
      const int r = rhalf + v;
      tile[r][col] = c[t][v] + bb + Xf[(size_t)(row0 + r) * DMODEL + col];
    }
  }
  __syncthreads();

  const int r    = threadIdx.x >> 4;
  const int cidx = threadIdx.x & 15;
  float s = 0.0f, ss = 0.0f;
#pragma unroll 8
  for (int i = 0; i < DMODEL / 16; ++i) {
    const float x = tile[r][cidx + 16 * i];
    s += x; ss += x * x;
  }
  s  += __shfl_xor(s, 1, 16);  s  += __shfl_xor(s, 2, 16);
  s  += __shfl_xor(s, 4, 16);  s  += __shfl_xor(s, 8, 16);
  ss += __shfl_xor(ss, 1, 16); ss += __shfl_xor(ss, 2, 16);
  ss += __shfl_xor(ss, 4, 16); ss += __shfl_xor(ss, 8, 16);

  const float mu  = s * (1.0f / DMODEL);
  const float var = ss * (1.0f / DMODEL) - mu * mu;
  const float rsg = rsqrtf(var + 1e-5f);

#pragma unroll 8
  for (int i = 0; i < DMODEL / 16; ++i) {
    const int col = cidx + 16 * i;
    const float x = tile[r][col];
    Out[(size_t)(row0 + r) * DMODEL + col] = (x - mu) * rsg * gamma[col] + beta[col];
  }
}

// ---------------------------------------------------------------------------
// Launch
// ---------------------------------------------------------------------------
extern "C" void kernel_launch(void* const* d_in, const int* in_sizes, int n_in,
                              void* d_out, int out_size, void* d_ws, size_t ws_size,
                              hipStream_t stream) {
  (void)in_sizes; (void)n_in; (void)out_size; (void)ws_size;
  const float* X    = (const float*)d_in[0];
  const float* mask = (const float*)d_in[1];
  const float* Wq   = (const float*)d_in[2];
  const float* bq   = (const float*)d_in[3];
  const float* Wk   = (const float*)d_in[4];
  const float* bk   = (const float*)d_in[5];
  const float* Wv   = (const float*)d_in[6];
  const float* bv   = (const float*)d_in[7];
  const float* Wo   = (const float*)d_in[8];
  const float* bo   = (const float*)d_in[9];
  const float* gam  = (const float*)d_in[10];
  const float* bet  = (const float*)d_in[11];
  float* out = (float*)d_out;

  char*  ws  = (char*)d_ws;
  size_t off = 0;
  auto alloc = [&](size_t bytes) -> void* {
    void* p = ws + off;
    off += (bytes + 255) & ~(size_t)255;
    return p;
  };
  __bf16* Xb  = (__bf16*)alloc((size_t)ROWS * DMODEL * 2);      // 16 MB
  __bf16* WqT = (__bf16*)alloc((size_t)DMODEL * DMODEL * 2);    //  2 MB
  __bf16* WkT = (__bf16*)alloc((size_t)DMODEL * DMODEL * 2);
  __bf16* WvT = (__bf16*)alloc((size_t)DMODEL * DMODEL * 2);
  __bf16* WoT = (__bf16*)alloc((size_t)DMODEL * DMODEL * 2);
  __bf16* Qb  = (__bf16*)alloc((size_t)ROWS * DMODEL * 2);      // 16 MB
  __bf16* Kb  = (__bf16*)alloc((size_t)ROWS * DMODEL * 2);
  __bf16* Vt  = (__bf16*)alloc((size_t)ROWS * DMODEL * 2);
  __bf16* Ctx = (__bf16*)alloc((size_t)ROWS * DMODEL * 2);      // total ~88 MB

  const int nX = ROWS * DMODEL;          // 8388608
  const int nW = DMODEL * DMODEL;        // 1048576

  cvt_kernel<<<(nX + 255) / 256, 256, 0, stream>>>(X, Xb, nX);
  tcvt_kernel<<<nW / 256, 256, 0, stream>>>(Wq, WqT);
  tcvt_kernel<<<nW / 256, 256, 0, stream>>>(Wk, WkT);
  tcvt_kernel<<<nW / 256, 256, 0, stream>>>(Wv, WvT);
  tcvt_kernel<<<nW / 256, 256, 0, stream>>>(Wo, WoT);

  qkv_kernel<<<dim3(ROWS / 16, 24), 256, 0, stream>>>(
      Xb, WqT, WkT, WvT, bq, bk, bv, Qb, Kb, Vt);

  attn_kernel<<<(BATCH * HEADS * SEQ / 16) / 8, 256, 0, stream>>>(
      Qb, Kb, Vt, mask, Ctx);

  oproj_ln_kernel<<<ROWS / 16, 256, 0, stream>>>(
      Ctx, WoT, bo, X, gam, bet, out);
}
